// GlobalGraph_51977694216713
// MI455X (gfx1250) — matrix operand order repaired and checked by reading, not compiled, not run on hardware
//
#include <hip/hip_runtime.h>
#include <hip/hip_bf16.h>

// ---------------------------------------------------------------------------
// BERT-style single-head attention for MI455X (gfx1250, wave32, WMMA).
//   B=64, S=1024, HID=DH=128, scale = 1/sqrt(128), mask bias = (1-m)*-1e4
// Pipeline:
//   kernel 0: convert Wq/Wk/Wv f32 -> f16 (one-shot, tiny)
//   kernel 1: Q = (X Wq^T + bq)*scale ; K = X Wk^T ; V^T = (X Wv^T + bv)^T
//             X tile staged through LDS as f16 once per block; outputs f16
//             into workspace (L2-resident, ~50 MB < 192 MB L2)
//   kernel 2: per (batch, 64-row q tile): scores = Q K^T + maskbias -> LDS
//             f16 (64x1032 strip, ~132 KB LDS), 8-wide vectorized in-place
//             softmax with deferred 1/sum, ctx = P V with V fragments
//             hoisted over the 4 q-subtiles. Mask loads / out stores are
//             non-temporal so the 268 MB mask stream does not evict
//             L2-resident K/V.
// All matmuls use v_wmma_f32_16x16x32_f16.
// ---------------------------------------------------------------------------

typedef __attribute__((ext_vector_type(16))) _Float16 v16h;
typedef __attribute__((ext_vector_type(8)))  _Float16 v8h;
typedef __attribute__((ext_vector_type(8)))  float    v8f;

#define BATCH 64
#define SEQ   1024
#define HID   128
#define MROWS (BATCH * SEQ)          // 65536
#define SCALE 0.08838834764831845f   // 1/sqrt(128)
#define QROWS 64                     // q rows per attention block (4 tiles)

__device__ __forceinline__ v8f wmma16x16x32(v16h a, v16h b, v8f c) {
  return __builtin_amdgcn_wmma_f32_16x16x32_f16(
      /*neg_a=*/false, a, /*neg_b=*/false, b,
      /*c_mod=*/(short)0, c, /*reuse_a=*/false, /*reuse_b=*/false);
}

// K-index inside an A fragment (16x32 f16, contiguous k = 16g + e, matching B): element e of v16h.
__device__ __forceinline__ int a_frag_k(int e, int g) {
  return 16 * g + e;
}

// A fragment from row-major f16 source (stride ld), tile origin (row0, k0).
// Per lane this is two contiguous 16B reads -> b128 loads.
__device__ __forceinline__ v16h load_a_f16(const _Float16* __restrict__ base,
                                           int ld, long row0, int k0) {
  int lane = threadIdx.x & 31;
  int m = lane & 15, g = lane >> 4;
  const _Float16* p = base + (row0 + m) * (long)ld + k0;
  v16h a;
#pragma unroll
  for (int e = 0; e < 16; ++e) a[e] = p[a_frag_k(e, g)];
  return a;
}

// B fragment (32x16, B[k][n]) where B[k][n] = base[(row0+n)*ld + k0 + k].
// Contiguous 16 f16 per lane (one 32B read). Works for W^T (W is [n][k]
// row-major), for K^T (K row-major [s][d]) and for V^T stored [d][s].
__device__ __forceinline__ v16h load_b_rows_f16(const _Float16* __restrict__ base,
                                                int ld, long row0, int k0) {
  int lane = threadIdx.x & 31;
  int n = lane & 15, g = lane >> 4;
  const _Float16* p = base + (row0 + n) * (long)ld + k0 + 16 * g;
  v16h b;
#pragma unroll
  for (int e = 0; e < 16; ++e) b[e] = p[e];
  return b;
}

// ---------------------------------------------------------------------------
// Kernel 0: one-shot weight conversion f32 -> f16 (3 x 128x128).
// ---------------------------------------------------------------------------
__global__ __launch_bounds__(256) void cvt_weights_kernel(
    const float* __restrict__ Wq, const float* __restrict__ Wk,
    const float* __restrict__ Wv,
    _Float16* __restrict__ Wqh, _Float16* __restrict__ Wkh,
    _Float16* __restrict__ Wvh) {
  const int i = blockIdx.x * 256 + threadIdx.x;   // grid covers 16384
  if (i < HID * HID) {
    Wqh[i] = (_Float16)Wq[i];
    Wkh[i] = (_Float16)Wk[i];
    Wvh[i] = (_Float16)Wv[i];
  }
}

// ---------------------------------------------------------------------------
// Kernel 1: QKV projection. Block = 256 threads (8 waves), 32 rows per block.
// X tile converted once into LDS f16; wave w owns output cols [16w, 16w+16).
// ---------------------------------------------------------------------------
#define XS_STRIDE 136   // 128 + 8 halves pad

__global__ __launch_bounds__(256) void qkv_proj_kernel(
    const float* __restrict__ X,
    const float* __restrict__ bq, const float* __restrict__ bv,
    const _Float16* __restrict__ Wqh, const _Float16* __restrict__ Wkh,
    const _Float16* __restrict__ Wvh,
    _Float16* __restrict__ Qh,   // [MROWS][128], scale+bias folded in
    _Float16* __restrict__ Kh,   // [MROWS][128]
    _Float16* __restrict__ Vt) { // [B][128][S]
  __shared__ _Float16 xs[32][XS_STRIDE];

  const int tid = threadIdx.x;
  const int w = tid >> 5;
  const int lane = tid & 31;
  const int lane16 = lane & 15, g = lane >> 4;
  const long m0 = (long)blockIdx.x * 32;
  const int n0 = w * 16;

  // Cooperative load + convert of the 32x128 X tile (16 f32 per thread).
  {
    const int row = tid >> 3;            // 0..31
    const int col0 = (tid & 7) * 16;     // 0..112
    const float* p = X + (m0 + row) * (long)HID + col0;
#pragma unroll
    for (int e = 0; e < 16; ++e) xs[row][col0 + e] = (_Float16)p[e];
  }
  __syncthreads();

  v8f accq[2], acck[2], accv[2];
#pragma unroll
  for (int t = 0; t < 2; ++t) { accq[t] = {}; acck[t] = {}; accv[t] = {}; }

#pragma unroll
  for (int ks = 0; ks < 4; ++ks) {
    const int k0 = ks * 32;
    v16h a0, a1;
#pragma unroll
    for (int e = 0; e < 16; ++e) {
      const int kk = k0 + a_frag_k(e, g);
      a0[e] = xs[lane16][kk];
      a1[e] = xs[16 + lane16][kk];
    }
    v16h bwq = load_b_rows_f16(Wqh, HID, n0, k0);
    v16h bwk = load_b_rows_f16(Wkh, HID, n0, k0);
    v16h bwv = load_b_rows_f16(Wvh, HID, n0, k0);
    accq[0] = wmma16x16x32(a0, bwq, accq[0]);
    accq[1] = wmma16x16x32(a1, bwq, accq[1]);
    acck[0] = wmma16x16x32(a0, bwk, acck[0]);
    acck[1] = wmma16x16x32(a1, bwk, acck[1]);
    accv[0] = wmma16x16x32(a0, bwv, accv[0]);
    accv[1] = wmma16x16x32(a1, bwv, accv[1]);
  }

  const int n = n0 + lane16;
  const float biasq = bq[n];
  const float biasv = bv[n];
#pragma unroll
  for (int t = 0; t < 2; ++t) {
#pragma unroll
    for (int r = 0; r < 8; ++r) {
      const long m = m0 + t * 16 + r + 8 * g;   // global row in [0, MROWS)
      Qh[m * HID + n] = (_Float16)((accq[t][r] + biasq) * SCALE);
      Kh[m * HID + n] = (_Float16)acck[t][r];
      const long bidx = m >> 10;                 // batch
      const long s = m & 1023;                   // position
      Vt[bidx * (HID * (long)SEQ) + (long)n * SEQ + s] =
          (_Float16)(accv[t][r] + biasv);
    }
  }
}

// ---------------------------------------------------------------------------
// Kernel 2: attention. Block = 256 threads (8 waves) handles one
// (batch, 64-row q tile). Scores staged f16 in LDS (~132 KB); softmax in
// place (8-wide vectorized); normalization deferred as a reciprocal
// multiply at the context store. K fragments are reused across 4
// q-subtiles, V fragments across 4 accumulators, so the per-batch K/V
// stream is read from L2 only once per 64 q rows.
// ---------------------------------------------------------------------------
#define SC_STRIDE 1032   // 1024 + 8 halves pad; row stride 2064 B (16B align)

__global__ __launch_bounds__(256) void attn_kernel(
    const _Float16* __restrict__ Qh,
    const _Float16* __restrict__ Kh,
    const _Float16* __restrict__ Vt,
    const float* __restrict__ mask,   // [B][S][S]
    float* __restrict__ out) {        // [B][S][128]
  __shared__ _Float16 sc[QROWS][SC_STRIDE];   // ~132 KB
  __shared__ float red[QROWS][4];
  __shared__ float rowmax[QROWS];
  __shared__ float rowrcp[QROWS];             // 1 / rowsum

  const int tid = threadIdx.x;
  const int w = tid >> 5;
  const int lane = tid & 31;
  const int lane16 = lane & 15, g = lane >> 4;
  const int q0 = blockIdx.x * QROWS;
  const int bb = blockIdx.y;
  const long qrow0 = (long)bb * SEQ + q0;     // row index into [B*S]

  // ---- Phase 1: scores = Qs K^T + (1-mask)*NEG, stored f16 in LDS --------
  v16h qa[4][4];                               // 4 q-subtiles x 4 d-steps
#pragma unroll
  for (int qt = 0; qt < 4; ++qt)
#pragma unroll
    for (int d = 0; d < 4; ++d)
      qa[qt][d] = load_a_f16(Qh, HID, qrow0 + qt * 16, d * 32);

  for (int kt = w; kt < 64; kt += 8) {
    v16h kb[4];
#pragma unroll
    for (int d = 0; d < 4; ++d)
      kb[d] = load_b_rows_f16(Kh, HID, (long)bb * SEQ + kt * 16, d * 32);
    const int kcol = kt * 16 + lane16;
#pragma unroll
    for (int qt = 0; qt < 4; ++qt) {
      v8f acc = {};
#pragma unroll
      for (int d = 0; d < 4; ++d) acc = wmma16x16x32(qa[qt][d], kb[d], acc);
      const float* mcol = mask + (qrow0 + qt * 16) * (long)SEQ + kcol;
#pragma unroll
      for (int r = 0; r < 8; ++r) {
        const int m = r + 8 * g;
        const float mv = __builtin_nontemporal_load(mcol + (long)m * SEQ);
        sc[qt * 16 + m][kcol] = (_Float16)(acc[r] + (1.0f - mv) * -10000.0f);
      }
    }
  }
  __syncthreads();

  // ---- Phase 2: in-place softmax, 8-wide (4 threads per row) -------------
  // Each row's 1024 halves = 128 aligned v8h chunks; thread (row, c0) owns
  // chunks c0, c0+4, ... so all LDS traffic is ds_load/store_b128.
  const int row = tid >> 2;        // 0..63
  const int c0 = tid & 3;
  v8h* __restrict__ rowp = (v8h*)sc[row];

  _Float16 hmax = (_Float16)(-60000.0f);   // below any masked score
  for (int ch = c0; ch < 128; ch += 4) {
    const v8h v = rowp[ch];
#pragma unroll
    for (int e = 0; e < 8; ++e) hmax = (v[e] > hmax) ? v[e] : hmax;
  }
  red[row][c0] = (float)hmax;
  __syncthreads();
  if (c0 == 0)
    rowmax[row] = fmaxf(fmaxf(red[row][0], red[row][1]),
                        fmaxf(red[row][2], red[row][3]));
  __syncthreads();
  const float rm = rowmax[row];
  float sum = 0.0f;
  for (int ch = c0; ch < 128; ch += 4) {
    const v8h v = rowp[ch];
    v8h ev;
#pragma unroll
    for (int e = 0; e < 8; ++e) {
      const float x = __expf((float)v[e] - rm);
      sum += x;
      ev[e] = (_Float16)x;
    }
    rowp[ch] = ev;                // unnormalized P, f16, in place
  }
  red[row][c0] = sum;
  __syncthreads();
  if (c0 == 0)
    rowrcp[row] =
        1.0f / (red[row][0] + red[row][1] + red[row][2] + red[row][3]);
  __syncthreads();

  // ---- Phase 3: ctx = P V ; wave w owns output cols [16w, 16w+16) --------
  v8f acc[4];
#pragma unroll
  for (int qt = 0; qt < 4; ++qt) acc[qt] = {};
  const _Float16* vtb = Vt + (long)bb * (HID * (long)SEQ);
  for (int k0 = 0; k0 < SEQ; k0 += 32) {
    // B fragment: B[k][n=d] = Vt[b][d][k], contiguous per lane; load once,
    // reuse across the 4 q-subtiles.
    v16h vb = load_b_rows_f16(vtb, SEQ, (long)w * 16, k0);
#pragma unroll
    for (int qt = 0; qt < 4; ++qt) {
      v16h pa;
#pragma unroll
      for (int e = 0; e < 16; ++e)
        pa[e] = sc[qt * 16 + lane16][k0 + a_frag_k(e, g)];
      acc[qt] = wmma16x16x32(pa, vb, acc[qt]);
    }
  }

  const int n = w * 16 + lane16;
#pragma unroll
  for (int qt = 0; qt < 4; ++qt) {
#pragma unroll
    for (int r = 0; r < 8; ++r) {
      const int m = qt * 16 + r + 8 * g;
      const float v = acc[qt][r] * rowrcp[m];
      __builtin_nontemporal_store(v, out + (qrow0 + m) * (long)HID + n);
    }
  }
}

// ---------------------------------------------------------------------------
extern "C" void kernel_launch(void* const* d_in, const int* in_sizes, int n_in,
                              void* d_out, int out_size, void* d_ws, size_t ws_size,
                              hipStream_t stream) {
  const float* X    = (const float*)d_in[0];  // hidden_states [B,S,HID]
  const float* mask = (const float*)d_in[1];  // attention_mask [B,S,S]
  const float* Wq   = (const float*)d_in[2];
  const float* bq   = (const float*)d_in[3];
  const float* Wk   = (const float*)d_in[4];
  const float* Wv   = (const float*)d_in[5];
  const float* bv   = (const float*)d_in[6];
  float* out = (float*)d_out;

  _Float16* Qh  = (_Float16*)d_ws;                      // 16 MB
  _Float16* Kh  = Qh + (size_t)MROWS * HID;             // 16 MB
  _Float16* Vt  = Kh + (size_t)MROWS * HID;             // 16 MB
  _Float16* Wqh = Vt + (size_t)MROWS * HID;             // 32 KB
  _Float16* Wkh = Wqh + (size_t)HID * HID;
  _Float16* Wvh = Wkh + (size_t)HID * HID;

  cvt_weights_kernel<<<(HID * HID + 255) / 256, 256, 0, stream>>>(
      Wq, Wk, Wv, Wqh, Wkh, Wvh);
  qkv_proj_kernel<<<MROWS / 32, 256, 0, stream>>>(X, bq, bv, Wqh, Wkh, Wvh,
                                                  Qh, Kh, Vt);
  attn_kernel<<<dim3(SEQ / QROWS, BATCH), 256, 0, stream>>>(Qh, Kh, Vt, mask,
                                                            out);
}
